// SelfAttention_12790412608047
// MI455X (gfx1250) — compile-verified, matrix operand order checked by reading, not run on hardware
//
#include <hip/hip_runtime.h>
#include <math.h>

// ---------------------------------------------------------------------------
// Problem constants (reference: B,T,C,H = 2,2048,2048,16; D = 128)
// ---------------------------------------------------------------------------
#define B_  2
#define T_  2048
#define C_  2048
#define H_  16
#define D_  128
#define MT_ (B_ * T_)          // 4096 rows of x

typedef __bf16 bf16;
typedef __attribute__((ext_vector_type(8)))  bf16  bf16x8;
typedef __attribute__((ext_vector_type(16))) bf16  bf16x16;
typedef __attribute__((ext_vector_type(8)))  float f32x8;
typedef __attribute__((ext_vector_type(4)))  unsigned int u32x4;
typedef __attribute__((ext_vector_type(8)))  int i32x8;
typedef __attribute__((ext_vector_type(4)))  int i32x4;

// ---------------------------------------------------------------------------
// WMMA helpers (CDNA5 wave32, V_WMMA_F32_16X16X32_BF16)
//
// 16-bit A-fragment layout (ISA 7.12.2): lane L holds row m=L&15; the 16
// elements are two contiguous K-chunks of 8: K = half*8 + [0..7] and
// K = 16 + half*8 + [0..7], half = L>>4. The B-fragment is the same pattern
// with n = L&15 over an n-major (transposed) tile, so both fragments are two
// 16-byte contiguous loads per lane from a [row][k] tile with 16B-aligned
// row stride.
// ---------------------------------------------------------------------------
__device__ __forceinline__ bf16x16 load_frag(const bf16* base, int stride) {
  const int lane = threadIdx.x & 31;
  const int m    = lane & 15;
  const int half = lane >> 4;
  const bf16* p = base + m * stride + half * 8;
  bf16x8 lo = *(const bf16x8*)(p);
  bf16x8 hi = *(const bf16x8*)(p + 16);
  return __builtin_shufflevector(lo, hi, 0, 1, 2, 3, 4, 5, 6, 7,
                                 8, 9, 10, 11, 12, 13, 14, 15);
}

__device__ __forceinline__ f32x8 wmma_bf16(bf16x16 a, bf16x16 b, f32x8 c) {
  // (neg_a, A, neg_b, B, c_mod, C, reuse_a, reuse_b)
  return __builtin_amdgcn_wmma_f32_16x16x32_bf16(false, a, false, b,
                                                 (short)0, c, false, false);
}

// ---------------------------------------------------------------------------
// Tensor Data Mover: DMA a 2D tile (tile_d0 x tile_d1 elements, 2B each) from
// global [tensor_d1][tensor_d0_stride] into LDS at lds_byte_addr, with
// hardware row padding so LDS row pitch = tile_d0*2 + pad_amount_dwords*4.
// Descriptor bitfields per CDNA5 ISA 8.3/8.4 (D# groups 0/1; groups 2/3 and
// the trailing group are zero-filled: tile_dim2=0 marks the tile as 2D).
// This toolchain exposes the 6-arg builtin (g0, g1, g2, g3, g4, cpol).
// ---------------------------------------------------------------------------
__device__ __forceinline__ void tdm_load_tile_2d_bf16(
    unsigned lds_byte_addr, const void* gptr, unsigned tensor_d0,
    unsigned tensor_d1, unsigned tensor_d0_stride, unsigned tile_d0,
    unsigned tile_d1, unsigned pad_interval, unsigned pad_amount) {
  unsigned long long ga = (unsigned long long)(size_t)gptr;
  u32x4 g0;
  g0[0] = 1u;                                     // count=1, user descriptor
  g0[1] = lds_byte_addr;                          // lds_addr [63:32]
  g0[2] = (unsigned)(ga & 0xffffffffu);           // global_addr lo
  g0[3] = (unsigned)((ga >> 32) & 0x01ffffffu)    // global_addr hi (57b)
          | (2u << 30);                           // type=2 ("image")
  i32x8 g1;
  g1[0] = (int)((1u << 16)                        // data_size=1 -> 2 bytes
                | (1u << 20)                      // pad_enable
                | (pad_interval << 22)            // pad_interval
                | (pad_amount << 25));            // pad_amount
  g1[1] = (int)((tensor_d0 & 0xffffu) << 16);     // [63:48]=tensor_dim0.lo
  g1[2] = (int)((tensor_d0 >> 16) | ((tensor_d1 & 0xffffu) << 16));
  g1[3] = (int)((tensor_d1 >> 16) | (tile_d0 << 16));  // [127:112]=tile_dim0
  g1[4] = (int)(tile_d1 & 0xffffu);               // tile_dim1; tile_dim2=0
  g1[5] = (int)tensor_d0_stride;                  // dim0 stride lo32
  g1[6] = 0;                                      // stride hi / dim1 stride
  g1[7] = 0;
  i32x4 gz4 = {0, 0, 0, 0};
  i32x8 gz8 = {0, 0, 0, 0, 0, 0, 0, 0};
  __builtin_amdgcn_tensor_load_to_lds(g0, g1, gz4, gz4, gz8, 0);
}

// ---------------------------------------------------------------------------
// fp32 -> bf16 conversion (grid-stride)
// ---------------------------------------------------------------------------
__global__ void f2bf_kernel(const float* __restrict__ in,
                            bf16* __restrict__ out, int n) {
  int i = blockIdx.x * blockDim.x + threadIdx.x;
  const int stride = gridDim.x * blockDim.x;
  for (; i < n; i += stride) out[i] = (bf16)in[i];
}

// ---------------------------------------------------------------------------
// padding_mask (prefix-true booleans, 1 byte each) -> per-batch length
// ---------------------------------------------------------------------------
__global__ void len_kernel(const unsigned char* __restrict__ mask,
                           int* __restrict__ len) {
  int b = threadIdx.x;
  if (b < B_) {
    int c = 0;
    for (int t = 0; t < T_; ++t) c += (mask[b * T_ + t] != 0) ? 1 : 0;
    len[b] = c;
  }
}

// ---------------------------------------------------------------------------
// Block-tiled bf16 WMMA GEMM:  C[M][N] = A[M][K] * B[K][N]
// 256 threads = 8 waves, 128x128 block tile, K-step 32.
// A tile is DMA'd by the Tensor Data Mover into LDS row-major [128][40]
// (hardware padding gives the 80B row pitch). B is staged n-major [128][40]
// by the waves so B fragments load exactly like A fragments.
// ---------------------------------------------------------------------------
template <bool OUT_BF16>
__global__ __launch_bounds__(256) void gemm_kernel(
    const bf16* __restrict__ A, const bf16* __restrict__ Bm,
    void* __restrict__ Cout, int M, int N, int K) {
  __shared__ bf16 As[128 * 40];
  __shared__ bf16 Bs[128 * 40];

  const int tid   = threadIdx.x;
  const int wid   = tid >> 5;
  const int lane  = tid & 31;
  const int waveM = wid >> 2;      // 0..1 -> 64-row strip
  const int waveN = wid & 3;       // 0..3 -> 32-col strip
  const int blkN  = blockIdx.x;
  const int blkM  = blockIdx.y;

  const f32x8 fzero = {0.f, 0.f, 0.f, 0.f, 0.f, 0.f, 0.f, 0.f};
  f32x8 acc[4][2];
#pragma unroll
  for (int i = 0; i < 4; ++i)
#pragma unroll
    for (int j = 0; j < 2; ++j) acc[i][j] = fzero;

  const unsigned asLds = (unsigned)(size_t)(void*)&As[0];

  for (int k0 = 0; k0 < K; k0 += 32) {
    // --- A tile via TDM: 32 (k, contiguous) x 128 (rows), pad 64B+16B rows
    if (wid == 0) {
      tdm_load_tile_2d_bf16(asLds, A + (size_t)blkM * 128 * K + k0,
                            (unsigned)K, (unsigned)M, (unsigned)K,
                            /*tile_d0=*/32u, /*tile_d1=*/128u,
                            /*pad_interval=*/3u /*16 dwords*/,
                            /*pad_amount=*/3u /*4 dwords*/);
    }
    // --- stage B tile transposed to n-major: read coalesced along n,
    //     scatter 8 b16 stores along k
#pragma unroll
    for (int i = 0; i < 2; ++i) {
      int c  = tid + i * 256;              // 0..511
      int k  = c >> 4;                     // 0..31
      int n8 = (c & 15) * 8;               // 0..120
      bf16x8 v =
          *(const bf16x8*)(Bm + (size_t)(k0 + k) * N + blkN * 128 + n8);
#pragma unroll
      for (int j = 0; j < 8; ++j) Bs[(n8 + j) * 40 + k] = v[j];
    }
    // prefetch next B tile chunk while we compute this one
    if (k0 + 32 < K) {
      int k  = tid >> 4;
      int n8 = (tid & 15) * 8;
      __builtin_prefetch(Bm + (size_t)(k0 + 32 + k) * N + blkN * 128 + n8, 0,
                         3);
    }
    if (wid == 0) __builtin_amdgcn_s_wait_tensorcnt(0);
    __syncthreads();

    bf16x16 afrag[4], bfrag[2];
#pragma unroll
    for (int i = 0; i < 4; ++i)
      afrag[i] = load_frag(&As[(waveM * 64 + i * 16) * 40], 40);
#pragma unroll
    for (int j = 0; j < 2; ++j)
      bfrag[j] = load_frag(&Bs[(waveN * 32 + j * 16) * 40], 40);
#pragma unroll
    for (int i = 0; i < 4; ++i)
#pragma unroll
      for (int j = 0; j < 2; ++j)
        acc[i][j] = wmma_bf16(afrag[i], bfrag[j], acc[i][j]);
    __syncthreads();
  }

  // --- epilogue: C/D layout -> lane n = lane&15, VGPR v -> row half*8+v
  const int n    = lane & 15;
  const int half = lane >> 4;
#pragma unroll
  for (int i = 0; i < 4; ++i) {
#pragma unroll
    for (int j = 0; j < 2; ++j) {
      int row0 = blkM * 128 + waveM * 64 + i * 16 + half * 8;
      int col  = blkN * 128 + waveN * 32 + j * 16 + n;
#pragma unroll
      for (int v = 0; v < 8; ++v) {
        size_t idx = (size_t)(row0 + v) * N + col;
        if (OUT_BF16)
          ((bf16*)Cout)[idx] = (bf16)acc[i][j][v];
        else
          ((float*)Cout)[idx] = acc[i][j][v];
      }
    }
  }
}

// ---------------------------------------------------------------------------
// RoPE + head split. qkv is bf16 [B][T][3C]. Outputs:
//   Qr, Kr : bf16 [B][H][T][D]   (RoPE applied)   -> A/B frags contiguous in d
//   Vt     : bf16 [B][H][D][T]   (transposed)     -> P@V B-frags contiguous in t
// One thread per (b,t,h,j), j < D/2 handles the rotated pair (j, j+64).
// ---------------------------------------------------------------------------
__global__ void rope_split_kernel(const bf16* __restrict__ qkv,
                                  bf16* __restrict__ Qr,
                                  bf16* __restrict__ Kr,
                                  bf16* __restrict__ Vt) {
  int idx = blockIdx.x * blockDim.x + threadIdx.x;   // < B*T*H*64 = 2^22
  int j = idx & 63;
  int h = (idx >> 6) & (H_ - 1);
  int t = (idx >> 10) & (T_ - 1);
  int b = idx >> 21;

  const float LOG10000_OVER_64 = 0.14391156f;        // ln(10000)/64
  float invf = __expf(-(float)j * LOG10000_OVER_64);
  float ang  = (float)t * invf;
  float sn, cs;
  __sincosf(ang, &sn, &cs);

  size_t base = (size_t)(b * T_ + t) * (3 * C_);
  int off = h * D_ + j;
  float q1 = (float)qkv[base + off];
  float q2 = (float)qkv[base + off + 64];
  float k1 = (float)qkv[base + C_ + off];
  float k2 = (float)qkv[base + C_ + off + 64];
  float v1 = (float)qkv[base + 2 * C_ + off];
  float v2 = (float)qkv[base + 2 * C_ + off + 64];

  size_t qb = (size_t)((b * H_ + h) * T_ + t) * D_ + j;
  Qr[qb]      = (bf16)(q1 * cs - q2 * sn);
  Qr[qb + 64] = (bf16)(q2 * cs + q1 * sn);
  Kr[qb]      = (bf16)(k1 * cs - k2 * sn);
  Kr[qb + 64] = (bf16)(k2 * cs + k1 * sn);

  size_t vb = (size_t)((b * H_ + h) * D_ + j) * T_ + t;
  Vt[vb]                   = (bf16)v1;
  Vt[vb + (size_t)64 * T_] = (bf16)v2;
}

// ---------------------------------------------------------------------------
// Fused causal flash attention (bf16 WMMA, f32 accumulators).
// 8 waves / block; each wave owns one (b, h, 16-row query tile) and streams
// key strips of 32, keeping the 16x128 output and row stats in registers.
// P (probabilities) is bounced through a per-wave LDS tile to convert the
// WMMA C/D layout into an A-fragment layout for the P@V WMMA.
// Output written as bf16 [B][T][C] with padded rows zeroed (matches
// reference's nan_to_num + output mask).
// ---------------------------------------------------------------------------
__global__ __launch_bounds__(256) void attn_kernel(
    const bf16* __restrict__ Qr, const bf16* __restrict__ Kr,
    const bf16* __restrict__ Vt, const int* __restrict__ len,
    bf16* __restrict__ Obf) {
  __shared__ bf16 pst_all[8][16 * 40];

  const int tid  = threadIdx.x;
  const int wid  = tid >> 5;
  const int lane = tid & 31;
  const int n    = lane & 15;
  const int half = lane >> 4;

  int gid = blockIdx.x * 8 + wid;          // < B*H*(T/16) = 4096
  int b   = gid >> 11;
  int h   = (gid >> 7) & (H_ - 1);
  int qt  = gid & 127;
  int t0  = qt * 16;
  int L   = len[b];

  bf16* pst = pst_all[wid];
  size_t headQK = (size_t)(b * H_ + h) * T_ * D_;
  size_t headV  = (size_t)(b * H_ + h) * D_ * T_;

  if (t0 >= L) {  // fully padded tile -> zeros
#pragma unroll
    for (int d = 0; d < 8; ++d)
#pragma unroll
      for (int v = 0; v < 8; ++v)
        Obf[(size_t)(b * T_ + t0 + half * 8 + v) * C_ + h * D_ + d * 16 + n] =
            (bf16)0.f;
    return;
  }

  // Q fragments for the 16xD tile, resident for the whole kernel
  bf16x16 qf[4];
#pragma unroll
  for (int c = 0; c < 4; ++c)
    qf[c] = load_frag(Qr + headQK + (size_t)t0 * D_ + c * 32, D_);

  const f32x8 fzero = {0.f, 0.f, 0.f, 0.f, 0.f, 0.f, 0.f, 0.f};
  f32x8 o[8];
#pragma unroll
  for (int d = 0; d < 8; ++d) o[d] = fzero;
  float mrow[8], lrow[8];
#pragma unroll
  for (int v = 0; v < 8; ++v) { mrow[v] = -1e30f; lrow[v] = 0.f; }

  const float scale = 0.08838834764831845f;  // 1/sqrt(128)
  int s_hi = (L < t0 + 16) ? L : (t0 + 16);  // causal + padding horizon

  for (int s0 = 0; s0 < s_hi; s0 += 32) {
    // latency hiding: pull next strip's K/V lines toward the WGP
    if (s0 + 32 < s_hi) {
      __builtin_prefetch(Kr + headQK + (size_t)(s0 + 32 + n) * D_ + half * 64,
                         0, 3);
      __builtin_prefetch(Vt + headV + (size_t)(lane * 4) * T_ + s0 + 32, 0, 3);
    }

    // ---- S = Q * K^T for two 16x16 score tiles (cols s0.., s0+16..)
    f32x8 S0 = fzero, S1 = fzero;
#pragma unroll
    for (int c = 0; c < 4; ++c) {
      bf16x16 kf0 = load_frag(Kr + headQK + (size_t)s0 * D_ + c * 32, D_);
      S0 = wmma_bf16(qf[c], kf0, S0);
      bf16x16 kf1 = load_frag(Kr + headQK + (size_t)(s0 + 16) * D_ + c * 32, D_);
      S1 = wmma_bf16(qf[c], kf1, S1);
    }

    // ---- scale + causal/padding mask + strip row-max
    float p0[8], p1[8], smax[8];
#pragma unroll
    for (int v = 0; v < 8; ++v) {
      int t  = t0 + half * 8 + v;
      int sA = s0 + n;
      int sB = s0 + 16 + n;
      float a  = S0[v] * scale;
      float bb = S1[v] * scale;
      if (sA > t || sA >= L) a  = -1e30f;
      if (sB > t || sB >= L) bb = -1e30f;
      p0[v] = a;
      p1[v] = bb;
      smax[v] = fmaxf(a, bb);
    }
#pragma unroll
    for (int offm = 1; offm < 16; offm <<= 1)
#pragma unroll
      for (int v = 0; v < 8; ++v)
        smax[v] = fmaxf(smax[v], __shfl_xor(smax[v], offm, 32));

    // ---- online softmax update
    float alpha[8], rsum[8];
#pragma unroll
    for (int v = 0; v < 8; ++v) {
      float nm = fmaxf(mrow[v], smax[v]);
      alpha[v] = __expf(mrow[v] - nm);
      mrow[v]  = nm;
      p0[v] = __expf(p0[v] - nm);
      p1[v] = __expf(p1[v] - nm);
      rsum[v] = p0[v] + p1[v];
    }
#pragma unroll
    for (int offm = 1; offm < 16; offm <<= 1)
#pragma unroll
      for (int v = 0; v < 8; ++v)
        rsum[v] += __shfl_xor(rsum[v], offm, 32);
#pragma unroll
    for (int v = 0; v < 8; ++v) lrow[v] = lrow[v] * alpha[v] + rsum[v];
#pragma unroll
    for (int d = 0; d < 8; ++d)
#pragma unroll
      for (int v = 0; v < 8; ++v) o[d][v] *= alpha[v];

    // ---- stage P (C/D layout) into LDS as a row-major 16x32 bf16 tile
#pragma unroll
    for (int v = 0; v < 8; ++v) {
      int row = half * 8 + v;
      pst[row * 40 + n]      = (bf16)p0[v];
      pst[row * 40 + 16 + n] = (bf16)p1[v];
    }
    // cross-lane LDS handoff within the wave: drain DS pipe
    asm volatile("s_wait_dscnt 0" ::: "memory");
    bf16x16 pf = load_frag(pst, 40);

    // ---- O += P * V  (V^T layout makes B-frags contiguous along t)
#pragma unroll
    for (int d = 0; d < 8; ++d) {
      bf16x16 vf = load_frag(Vt + headV + (size_t)(d * 16) * T_ + s0, T_);
      o[d] = wmma_bf16(pf, vf, o[d]);
    }
  }

  // ---- normalize and write bf16 [B][T][C]; zero padded rows
  float inv[8];
#pragma unroll
  for (int v = 0; v < 8; ++v) {
    int t = t0 + half * 8 + v;
    inv[v] = (t < L && lrow[v] > 0.f) ? 1.f / lrow[v] : 0.f;
  }
#pragma unroll
  for (int d = 0; d < 8; ++d)
#pragma unroll
    for (int v = 0; v < 8; ++v)
      Obf[(size_t)(b * T_ + t0 + half * 8 + v) * C_ + h * D_ + d * 16 + n] =
          (bf16)(o[d][v] * inv[v]);
}

// ---------------------------------------------------------------------------
// Host-side orchestration (all on `stream`, graph-capture safe)
// ---------------------------------------------------------------------------
extern "C" void kernel_launch(void* const* d_in, const int* in_sizes, int n_in,
                              void* d_out, int out_size, void* d_ws,
                              size_t ws_size, hipStream_t stream) {
  const float* x = (const float*)d_in[0];
  const unsigned char* mask = (const unsigned char*)d_in[1];  // jnp.bool_ = 1B
  const float* Wqkv = (const float*)d_in[2];
  const float* Wout = (const float*)d_in[3];

  char* ws = (char*)d_ws;
  size_t off = 0;
  auto take = [&](size_t bytes) -> char* {
    char* p = ws + off;
    off += (bytes + 255) & ~(size_t)255;
    return p;
  };

  bf16* Xb    = (bf16*)take((size_t)MT_ * C_ * 2);          // 16.8 MB
  bf16* Wqkvb = (bf16*)take((size_t)C_ * 3 * C_ * 2);       // 25.2 MB
  bf16* Woutb = (bf16*)take((size_t)C_ * C_ * 2);           //  8.4 MB
  bf16* QKVb  = (bf16*)take((size_t)MT_ * 3 * C_ * 2);      // 50.3 MB
  bf16* Qr    = (bf16*)take((size_t)B_ * H_ * T_ * D_ * 2); // 16.8 MB
  bf16* Kr    = (bf16*)take((size_t)B_ * H_ * T_ * D_ * 2); // 16.8 MB
  bf16* Vt    = (bf16*)take((size_t)B_ * H_ * T_ * D_ * 2); // 16.8 MB
  bf16* Obf   = (bf16*)take((size_t)MT_ * C_ * 2);          // 16.8 MB
  int*  len   = (int*)take(256);

  // 1) fp32 -> bf16 operand conversion
  f2bf_kernel<<<4096, 256, 0, stream>>>(x, Xb, MT_ * C_);
  f2bf_kernel<<<4096, 256, 0, stream>>>(Wqkv, Wqkvb, C_ * 3 * C_);
  f2bf_kernel<<<4096, 256, 0, stream>>>(Wout, Woutb, C_ * C_);
  len_kernel<<<1, 32, 0, stream>>>(mask, len);

  // 2) QKV projection: [4096,2048] x [2048,6144] -> bf16 qkv
  gemm_kernel<true><<<dim3(3 * C_ / 128, MT_ / 128), 256, 0, stream>>>(
      Xb, Wqkvb, (void*)QKVb, MT_, 3 * C_, C_);

  // 3) RoPE + head split/transpose
  rope_split_kernel<<<(B_ * T_ * H_ * 64) / 256, 256, 0, stream>>>(QKVb, Qr,
                                                                   Kr, Vt);

  // 4) fused causal flash attention -> bf16 [B,T,C] (padded rows zeroed)
  attn_kernel<<<(B_ * H_ * (T_ / 16)) / 8, 256, 0, stream>>>(Qr, Kr, Vt, len,
                                                             Obf);

  // 5) output projection: [4096,2048] x [2048,2048] -> f32 d_out
  gemm_kernel<false><<<dim3(C_ / 128, MT_ / 128), 256, 0, stream>>>(
      Obf, Woutb, d_out, MT_, C_, C_);
}
